// NanoporeVQModel_3410204033662
// MI455X (gfx1250) — matrix operand order repaired and checked by reading, not compiled, not run on hardware
//
#include <hip/hip_runtime.h>
#include <hip/hip_bf16.h>
#include <math.h>

// ---------------------------------------------------------------------------
// NanoporeVQ model for MI455X (gfx1250, wave32).
//  - VQ codebook search (134-GFLOP GEMM) runs on v_wmma_f32_16x16x32_bf16
//    with a hi/lo bf16 split (3 WMMAs per 32-wide K step) for ~fp32 accuracy.
//  - Codebook tiles are staged into LDS by the Tensor Data Mover
//    (tensor_load_to_lds + TENSORcnt), one descriptor per 16-row half-tile,
//    double-buffered and shared by the 4 waves of the workgroup.
//  - Conv stages are bandwidth-bound (~300 MB total @ 23.3 TB/s) -> scalar VALU.
// ---------------------------------------------------------------------------

typedef __attribute__((ext_vector_type(16))) __bf16 v16bf;
typedef __attribute__((ext_vector_type(8)))  __bf16 v8bf;
typedef __attribute__((ext_vector_type(8)))  float  v8f;
typedef __attribute__((ext_vector_type(4)))  int    v4i;
typedef __attribute__((ext_vector_type(8)))  int    v8i;

#define B_ 8
#define L_ 20000
#define T_ 4000
#define D_ 256
#define K_ 8192
#define N_ (B_ * T_)          // 32000 VQ rows

#define ROWB_ 512             // codebook row bytes (256 * bf16)
#define LDSROWB_ 528          // padded LDS row stride (TDM pad: +4 dwords/row)
#define TILEB_ (16 * LDSROWB_)  // one 16-row tile in LDS (hi or lo): 8448 B

__device__ __forceinline__ float gelu_f(float x) {
  // exact (erf-based) GELU, matching jax.nn.gelu(approximate=False)
  return 0.5f * x * (1.0f + erff(x * 0.70710678118654752440f));
}

// ---- TDM: one descriptor moves a 16-row x 512B tile into LDS, padding each
// 128-dword row with 4 dwords (LDS row stride 528B). D# per ISA 08_async_tensor:
//  g0: count=1 | lds_addr[63:32] | global_addr[120:64] | type=2[127:126]
//  g1: data_size=4B[17:16], pad_enable[20], pad_interval=6(128dw)[24:22],
//      pad_amount=3(4dw)[31:25], tensor_dim0=128[79:48], tensor_dim1=16[111:80],
//      tile_dim0=128[127:112], tile_dim1=16[143:128], dim0_stride=128[207:160]
__device__ __forceinline__ void tdm_load_tile(uint64_t ga, uint32_t la) {
  v4i g0;
  g0.x = 1;                                           // count=1 (valid D#)
  g0.y = (int)la;                                     // lds_addr
  g0.z = (int)(uint32_t)ga;                           // global_addr[31:0]
  g0.w = (int)(((uint32_t)(ga >> 32)) & 0x01ffffffu)  // global_addr[56:32]
         | (int)0x80000000u;                          // type=2 ("image")
  v8i g1;
  g1.s0 = 0x07920000;   // data_size=4B, pad_enable, pad_interval=128dw, pad=4dw
  g1.s1 = 0x00800000;   // tensor_dim0[15:0]=128 @ bits 63:48
  g1.s2 = 0x00100000;   // tensor_dim1[15:0]=16  @ bits 95:80
  g1.s3 = 0x00800000;   // tile_dim0=128         @ bits 127:112
  g1.s4 = 16;           // tile_dim1=16          @ bits 143:128
  g1.s5 = 128;          // tensor_dim0_stride[31:0]=128 dwords
  g1.s6 = 0;
  g1.s7 = 0;
  asm volatile("tensor_load_to_lds %0, %1" :: "s"(g0), "s"(g1) : "memory");
}

// --------------------------- codebook prep ---------------------------------
__global__ __launch_bounds__(256) void prep_codebook(
    const float* __restrict__ cb, __bf16* __restrict__ ch,
    __bf16* __restrict__ cl, float* __restrict__ cnorm) {
  __shared__ float red[256];
  const int k = blockIdx.x, d = threadIdx.x;
  const float c = cb[(size_t)k * D_ + d];
  const __bf16 hi = (__bf16)c;
  ch[(size_t)k * D_ + d] = hi;
  cl[(size_t)k * D_ + d] = (__bf16)(c - (float)hi);
  red[d] = c * c;
  __syncthreads();
  for (int s = 128; s > 0; s >>= 1) {
    if (d < s) red[d] += red[d + s];
    __syncthreads();
  }
  if (d == 0) cnorm[k] = red[0];
}

// ------------------------------ encoder ------------------------------------
__global__ __launch_bounds__(256) void enc_conv1(
    const float* __restrict__ x, const float* __restrict__ w,
    float* __restrict__ out) {          // [8,1,L] -> [8,64,L], k9 p4, gelu
  const int gid = blockIdx.x * 256 + threadIdx.x;
  if (gid >= B_ * 64 * L_) return;
  const int l  = gid % L_;
  const int co = (gid / L_) % 64;
  const int b  = gid / (64 * L_);
  const float* xr = x + (size_t)b * L_;
  const float* wr = w + co * 9;
  float s = 0.f;
#pragma unroll
  for (int k = 0; k < 9; ++k) {
    const int p = l + k - 4;
    if (p >= 0 && p < L_) s += xr[p] * wr[k];
  }
  out[gid] = gelu_f(s);
}

__global__ __launch_bounds__(256) void enc_conv2(
    const float* __restrict__ h, const float* __restrict__ w,
    float* __restrict__ out) {          // [8,64,L] -> [8,128,T], k5 s5, gelu
  const int gid = blockIdx.x * 256 + threadIdx.x;
  if (gid >= B_ * 128 * T_) return;
  const int t  = gid % T_;
  const int co = (gid / T_) % 128;
  const int b  = gid / (128 * T_);
  const float* hb = h + (size_t)b * 64 * L_;
  const float* wr = w + (size_t)co * 64 * 5;
  float s = 0.f;
  for (int ci = 0; ci < 64; ++ci) {
    const float* hr = hb + (size_t)ci * L_ + 5 * t;
#pragma unroll
    for (int k = 0; k < 5; ++k) s += hr[k] * wr[ci * 5 + k];
  }
  out[gid] = gelu_f(s);
}

__global__ __launch_bounds__(256) void enc_conv3(
    const float* __restrict__ h, const float* __restrict__ w,
    float* __restrict__ zf, __bf16* __restrict__ zh, __bf16* __restrict__ zl) {
  // [8,128,T] -> z [8,256,T]; emitted transposed as zf[N,D] f32 + bf16 hi/lo
  const int gid = blockIdx.x * 256 + threadIdx.x;
  if (gid >= B_ * 256 * T_) return;
  const int t  = gid % T_;
  const int co = (gid / T_) % 256;
  const int b  = gid / (256 * T_);
  const float* hb = h + (size_t)b * 128 * T_;
  const float* wr = w + (size_t)co * 128 * 5;
  float s = 0.f;
  for (int ci = 0; ci < 128; ++ci) {
#pragma unroll
    for (int k = 0; k < 5; ++k) {
      const int p = t + k - 2;
      if (p >= 0 && p < T_) s += hb[(size_t)ci * T_ + p] * wr[ci * 5 + k];
    }
  }
  const int n = b * T_ + t;
  zf[(size_t)n * D_ + co] = s;
  const __bf16 hi = (__bf16)s;
  zh[(size_t)n * D_ + co] = hi;
  zl[(size_t)n * D_ + co] = (__bf16)(s - (float)hi);
}

// ------------------------- VQ argmin via WMMA ------------------------------
// Block = 4 waves. Each wave owns 16 z rows (A frags pinned in 128 VGPRs).
// Wave 0 drives the TDM: two tensor_load_to_lds per 16-codebook-row tile
// (hi+lo bf16), double-buffered; all 4 waves consume via ds_load -> WMMA.
// argmin target: dist(n) = ||c_n||^2 - 2<z,c_n>  (||z||^2 is row-constant).
__global__ __launch_bounds__(128) void vq_kernel(
    const __bf16* __restrict__ zh, const __bf16* __restrict__ zl,
    const __bf16* __restrict__ ch, const __bf16* __restrict__ cl,
    const float* __restrict__ cnorm, int* __restrict__ out_idx) {
  // [buf][hi/lo][16 rows * 528B]  -> 2*2*8448 = 33792 B of LDS
  __shared__ unsigned char lds_tile[2][2][TILEB_];

  const int tid  = threadIdx.x;       // 0..127
  const int lane = tid & 31;
  const int wave = tid >> 5;
  const int m0   = (blockIdx.x * 4 + wave) * 16;  // 500 blks * 4 waves * 16 = 32000
  const int hgrp = lane >> 4;                     // lane half (0 / 1)
  const int l16  = lane & 15;

  const uint32_t lds0 = (uint32_t)(size_t)&lds_tile[0][0][0];
  const uint64_t gch  = (uint64_t)(size_t)ch;
  const uint64_t gcl  = (uint64_t)(size_t)cl;

  // ---- A fragments: 16x32 bf16 A in this kernel's contiguous k order (B matches).
  // lanes 0-15 : row M=lane,    K runs {d0..d0+15}
  // lanes 16-31: row M=lane-16, K runs {d0+16..d0+31}
  v16bf a_hi[8], a_lo[8];
  const __bf16* zhrow = zh + (size_t)(m0 + l16) * D_;
  const __bf16* zlrow = zl + (size_t)(m0 + l16) * D_;
#pragma unroll
  for (int s = 0; s < 8; ++s) {
    const int k1 = s * 32 + (hgrp ? 16 : 0);
    const int k2 = s * 32 + (hgrp ? 24 : 8);
    v8bf h0  = *(const v8bf*)(zhrow + k1);
    v8bf h1v = *(const v8bf*)(zhrow + k2);
    v8bf l0  = *(const v8bf*)(zlrow + k1);
    v8bf l1v = *(const v8bf*)(zlrow + k2);
    a_hi[s] = __builtin_shufflevector(h0, h1v, 0, 1, 2, 3, 4, 5, 6, 7,
                                      8, 9, 10, 11, 12, 13, 14, 15);
    a_lo[s] = __builtin_shufflevector(l0, l1v, 0, 1, 2, 3, 4, 5, 6, 7,
                                      8, 9, 10, 11, 12, 13, 14, 15);
  }

  float minv[8];
  int   mini[8];
#pragma unroll
  for (int r = 0; r < 8; ++r) { minv[r] = 3.402823466e38f; mini[r] = 0; }

  // prologue: wave 0 stages tile 0 into buffer 0 via TDM
  if (wave == 0) {
    tdm_load_tile(gch, lds0);
    tdm_load_tile(gcl, lds0 + TILEB_);
  }

  const int NT = K_ / 16;  // 512 tiles
  for (int i = 0; i < NT; ++i) {
    const int buf = i & 1;
    // producer wave: TDM writes for tile i landed in LDS -> publish to block
    if (wave == 0) __builtin_amdgcn_s_wait_tensorcnt(0);
    __syncthreads();
    // stage tile i+1 into the other buffer while everyone computes on tile i
    if (wave == 0 && i + 1 < NT) {
      const uint64_t gadd = (uint64_t)(i + 1) * 16 * ROWB_;
      const uint32_t lbuf = lds0 + (uint32_t)((i + 1) & 1) * (2 * TILEB_);
      tdm_load_tile(gch + gadd, lbuf);
      tdm_load_tile(gcl + gadd, lbuf + TILEB_);
    }

    // B fragment: 32x16 bf16 B, same contiguous k order as A. Column N = lane&15;
    // lanes 0-15 carry K=d0..d0+15, lanes 16-31 carry K=d0+16..d0+31:
    // one contiguous 16-bf16 run per lane (two 16B LDS reads).
    const unsigned char* pch = &lds_tile[buf][0][l16 * LDSROWB_ + hgrp * 32];
    const unsigned char* pcl = &lds_tile[buf][1][l16 * LDSROWB_ + hgrp * 32];
    v8f acc = {};
#pragma unroll
    for (int s = 0; s < 8; ++s) {
      v8bf bh0 = *(const v8bf*)(pch + s * 64);
      v8bf bh1 = *(const v8bf*)(pch + s * 64 + 16);
      v8bf bl0 = *(const v8bf*)(pcl + s * 64);
      v8bf bl1 = *(const v8bf*)(pcl + s * 64 + 16);
      v16bf bh = __builtin_shufflevector(bh0, bh1, 0, 1, 2, 3, 4, 5, 6, 7,
                                         8, 9, 10, 11, 12, 13, 14, 15);
      v16bf bl = __builtin_shufflevector(bl0, bl1, 0, 1, 2, 3, 4, 5, 6, 7,
                                         8, 9, 10, 11, 12, 13, 14, 15);
      // split-precision: z_hi*c_hi + z_hi*c_lo + z_lo*c_hi (f32 accumulate)
      acc = __builtin_amdgcn_wmma_f32_16x16x32_bf16(false, a_hi[s], false, bh,
                                                    (short)0, acc, false, false);
      acc = __builtin_amdgcn_wmma_f32_16x16x32_bf16(false, a_hi[s], false, bl,
                                                    (short)0, acc, false, false);
      acc = __builtin_amdgcn_wmma_f32_16x16x32_bf16(false, a_lo[s], false, bh,
                                                    (short)0, acc, false, false);
    }
    const float cn = cnorm[i * 16 + l16];
#pragma unroll
    for (int r = 0; r < 8; ++r) {
      const float dist = cn - 2.0f * acc[r];
      if (dist < minv[r]) { minv[r] = dist; mini[r] = i * 16 + l16; }
    }
  }

  // C/D layout: lane holds column n=lane&15 for rows (hgrp?8:0)+r.
  // Reduce min across the 16 lanes of each half (xor masks stay in-half).
#pragma unroll
  for (int r = 0; r < 8; ++r) {
    float v = minv[r];
    int   i = mini[r];
    for (int m = 8; m >= 1; m >>= 1) {
      const float ov = __shfl_xor(v, m, 32);
      const int   oi = __shfl_xor(i, m, 32);
      if (ov < v || (ov == v && oi < i)) { v = ov; i = oi; }
    }
    if (l16 == 0) out_idx[m0 + hgrp * 8 + r] = i;
  }
}

// --------------------- gather + commit loss partials -----------------------
__global__ __launch_bounds__(256) void vq_gather(
    const int* __restrict__ idx, const float* __restrict__ cb,
    const float* __restrict__ zf, float* __restrict__ qt,
    float* __restrict__ partials, int* __restrict__ idx_out) {
  __shared__ float red[256];
  const int gid = blockIdx.x * 256 + threadIdx.x;   // exactly N_*D_ threads
  const int d = gid % D_;
  const int n = gid / D_;
  const int code = idx[n];
  const float q = cb[(size_t)code * D_ + d];
  const float diff = zf[gid] - q;
  const int b = n / T_, t = n % T_;
  qt[((size_t)b * D_ + d) * T_ + t] = q;            // straight-through fwd = q
  if (d == 0) idx_out[n] = code;
  red[threadIdx.x] = diff * diff;
  __syncthreads();
  for (int s = 128; s > 0; s >>= 1) {
    if (threadIdx.x < s) red[threadIdx.x] += red[threadIdx.x + s];
    __syncthreads();
  }
  if (threadIdx.x == 0) partials[blockIdx.x] = red[0];
}

__global__ __launch_bounds__(256) void loss_reduce(
    const float* __restrict__ partials, int nparts, float* __restrict__ out_loss) {
  __shared__ float red[256];
  float s = 0.f;
  for (int i = threadIdx.x; i < nparts; i += 256) s += partials[i];
  red[threadIdx.x] = s;
  __syncthreads();
  for (int t = 128; t > 0; t >>= 1) {
    if (threadIdx.x < t) red[threadIdx.x] += red[threadIdx.x + t];
    __syncthreads();
  }
  if (threadIdx.x == 0)
    out_loss[0] = red[0] / (float)((size_t)N_ * D_);  // COMMITMENT_WEIGHT = 1
}

// ------------------------------ decoder ------------------------------------
__global__ __launch_bounds__(256) void dec_conv1(
    const float* __restrict__ qt, const float* __restrict__ w,
    float* __restrict__ out) {          // [8,256,T] -> [8,128,T], k5 p2, gelu
  const int gid = blockIdx.x * 256 + threadIdx.x;
  if (gid >= B_ * 128 * T_) return;
  const int t  = gid % T_;
  const int co = (gid / T_) % 128;
  const int b  = gid / (128 * T_);
  const float* qb = qt + (size_t)b * 256 * T_;
  const float* wr = w + (size_t)co * 256 * 5;
  float s = 0.f;
  for (int ci = 0; ci < 256; ++ci) {
#pragma unroll
    for (int k = 0; k < 5; ++k) {
      const int p = t + k - 2;
      if (p >= 0 && p < T_) s += qb[(size_t)ci * T_ + p] * wr[ci * 5 + k];
    }
  }
  out[gid] = gelu_f(s);
}

__global__ __launch_bounds__(256) void dec_convT(
    const float* __restrict__ h, const float* __restrict__ w,
    float* __restrict__ out) {  // convT k5 s5 VALID: l -> (t=l/5, k=l%5), gelu
  const int gid = blockIdx.x * 256 + threadIdx.x;
  if (gid >= B_ * 64 * L_) return;
  const int l  = gid % L_;
  const int co = (gid / L_) % 64;
  const int b  = gid / (64 * L_);
  const int t = l / 5, k = l % 5;
  const float* hb = h + (size_t)b * 128 * T_ + t;
  float s = 0.f;
  for (int ci = 0; ci < 128; ++ci)
    s += hb[(size_t)ci * T_] * w[((size_t)ci * 64 + co) * 5 + k];  // IOH weights
  out[gid] = gelu_f(s);
}

__global__ __launch_bounds__(256) void dec_conv3(
    const float* __restrict__ h, const float* __restrict__ w,
    float* __restrict__ recon) {        // [8,64,L] -> [8,1,L], k9 p4
  const int gid = blockIdx.x * 256 + threadIdx.x;
  if (gid >= B_ * L_) return;
  const int l = gid % L_;
  const int b = gid / L_;
  const float* hb = h + (size_t)b * 64 * L_;
  float s = 0.f;
  for (int ci = 0; ci < 64; ++ci) {
#pragma unroll
    for (int k = 0; k < 9; ++k) {
      const int p = l + k - 4;
      if (p >= 0 && p < L_) s += hb[(size_t)ci * L_ + p] * w[ci * 9 + k];
    }
  }
  recon[gid] = s;
}

// ------------------------------- driver ------------------------------------
extern "C" void kernel_launch(void* const* d_in, const int* in_sizes, int n_in,
                              void* d_out, int out_size, void* d_ws, size_t ws_size,
                              hipStream_t stream) {
  const float* x   = (const float*)d_in[0];
  const float* we1 = (const float*)d_in[1];
  const float* we2 = (const float*)d_in[2];
  const float* we3 = (const float*)d_in[3];
  const float* cb  = (const float*)d_in[4];
  const float* wd1 = (const float*)d_in[5];
  const float* wd2 = (const float*)d_in[6];
  const float* wd3 = (const float*)d_in[7];
  (void)in_sizes; (void)n_in; (void)out_size; (void)ws_size;

  char* wsp = (char*)d_ws;
  size_t off = 0;
  auto alloc = [&](size_t bytes) -> char* {
    char* p = wsp + off;
    off += (bytes + 255) & ~(size_t)255;
    return p;
  };

  __bf16* cb_hi = (__bf16*)alloc((size_t)K_ * D_ * 2);   //  4.19 MB
  __bf16* cb_lo = (__bf16*)alloc((size_t)K_ * D_ * 2);   //  4.19 MB
  float*  cnorm = (float*) alloc((size_t)K_ * 4);
  float*  h1    = (float*) alloc((size_t)B_ * 64 * L_ * 4);   // 40.96 MB (reused for hd2)
  float*  h2    = (float*) alloc((size_t)B_ * 128 * T_ * 4);  // 16.38 MB
  float*  zf    = (float*) alloc((size_t)N_ * D_ * 4);        // 32.77 MB
  __bf16* zh    = (__bf16*)alloc((size_t)N_ * D_ * 2);        // 16.38 MB
  __bf16* zl    = (__bf16*)alloc((size_t)N_ * D_ * 2);        // 16.38 MB
  int*    idx   = (int*)   alloc((size_t)N_ * 4);
  float*  qt    = (float*) alloc((size_t)B_ * D_ * T_ * 4);   // 32.77 MB
  float*  hd1   = (float*) alloc((size_t)B_ * 128 * T_ * 4);  // 16.38 MB
  float*  parts = (float*) alloc((size_t)(N_ * D_ / 256) * 4);
  float*  hd2   = h1;  // decoder convT output reuses enc conv1 buffer

  float* outf     = (float*)d_out;                 // recon [8*20000]
  int*   idx_out  = (int*)d_out + B_ * L_;         // idx   [32000] (int32 bits)
  float* loss_out = outf + B_ * L_ + N_;           // commit_loss [1]

  prep_codebook<<<K_, 256, 0, stream>>>(cb, cb_hi, cb_lo, cnorm);
  enc_conv1<<<(B_ * 64 * L_ + 255) / 256, 256, 0, stream>>>(x, we1, h1);
  enc_conv2<<<(B_ * 128 * T_ + 255) / 256, 256, 0, stream>>>(h1, we2, h2);
  enc_conv3<<<(B_ * 256 * T_ + 255) / 256, 256, 0, stream>>>(h2, we3, zf, zh, zl);
  vq_kernel<<<N_ / 64, 128, 0, stream>>>(zh, zl, cb_hi, cb_lo, cnorm, idx);
  vq_gather<<<N_ * D_ / 256, 256, 0, stream>>>(idx, cb, zf, qt, parts, idx_out);
  loss_reduce<<<1, 256, 0, stream>>>(parts, N_ * D_ / 256, loss_out);
  dec_conv1<<<(B_ * 128 * T_ + 255) / 256, 256, 0, stream>>>(qt, wd1, hd1);
  dec_convT<<<(B_ * 64 * L_ + 255) / 256, 256, 0, stream>>>(hd1, wd2, hd2);
  dec_conv3<<<(B_ * L_ + 255) / 256, 256, 0, stream>>>(hd2, wd3, outf);
}
